// RAFT_9380208575055
// MI455X (gfx1250) — matrix operand order repaired and checked by reading, not compiled, not run on hardware
//
#include <hip/hip_runtime.h>
#include <math.h>

// ---------------- CDNA5 WMMA types ----------------
typedef __attribute__((ext_vector_type(16))) _Float16 v16h;
typedef __attribute__((ext_vector_type(8)))  _Float16 v8h;
typedef __attribute__((ext_vector_type(8)))  float    v8f;

#define H_     48
#define W_     64
#define HW_    (H_ * W_)        // 3072
#define NB_    2
#define NPIX_  (NB_ * HW_)      // 6144
#define DFEAT_ 256
#define ITERS_ 12

// activation codes
#define ACT_NONE 0
#define ACT_RELU 1
#define ACT_SIG  2
#define ACT_TANH 3
#define ACT_QRT  4

// =====================================================================
// Implicit-GEMM conv via v_wmma_f32_16x16x32_f16, LDS-staged.
// Block = 256 threads = 8 waves; tile M=64 (out ch) x N=64 (pixels).
// Per 32-K chunk:
//   A: 64x32 weights -> LDS, float2-vectorized loads, 1 ds_store_b128/thr
//   B: 64x32 im2col  -> LDS, batched scalar gathers,  1 ds_store_b128/thr
// Each wave: 1 A fragment + 2 B fragments -> 2 WMMAs per chunk.
// All gathers are branch-free (clamped address * valid mask); remaining
// guards are wave-uniform.
// =====================================================================
__global__ __launch_bounds__(256) void conv2d_wmma_f16(
    const float* __restrict__ wgt, const float* __restrict__ bias,
    const float* __restrict__ s0, int c0,
    const float* __restrict__ s1, int c1,
    const float* __restrict__ s2, int c2,
    const float* __restrict__ s3, int c3,
    float* __restrict__ out, int Cout, int KH, int KW, int act)
{
    __shared__ __attribute__((aligned(32))) _Float16 ldsA[64 * 32];  // [m][k]
    __shared__ __attribute__((aligned(32))) _Float16 ldsB[64 * 32];  // [px][k]

    const int t      = threadIdx.x;
    const int lane   = t & 31;
    const int wv     = t >> 5;
    const int mSub   = wv >> 1;          // 0..3 : 16-row slab of M
    const int nSub   = wv & 1;           // 0..1 : 32-px slab of N
    const int laneHi = lane >> 4;
    const int mn     = lane & 15;

    const int mBase = blockIdx.y * 64;
    const int nBase = blockIdx.x * 64;   // HW_ % 64 == 0 -> never crosses batch

    const int khw  = KH * KW;
    const int Ktot = (c0 + c1 + c2 + c3) * khw;
    const int ph   = KH >> 1, pw = KW >> 1;

    // ---- A-fill mapping: thread owns row aMl, 8 consecutive k at aK
    const int   aMl = t >> 2;            // 0..63
    const int   aK  = (t & 3) * 8;       // 0,8,16,24
    const int   mg  = mBase + aMl;
    const int   mc  = (mg < Cout) ? mg : (Cout - 1);
    const float fm  = (mg < Cout) ? 1.0f : 0.0f;

    // ---- B-fill mapping: thread owns pixel bPx, 8 consecutive k at bK
    const int bPx  = t & 63;
    const int bK   = (t >> 6) * 8;       // wave-uniform
    const int gpix = nBase + bPx;
    const int nImg = gpix / HW_;
    const int ppix = gpix - nImg * HW_;
    const int pyy  = ppix / W_;
    const int pxx  = ppix - pyy * W_;

    v8f acc[2] = {};
    for (int k0 = 0; k0 < Ktot; k0 += 32) {
        __syncthreads();   // protect previous chunk's fragment reads

        // ================= stage A (weights) =================
        {
            float tmpA[8];
            const float* wr = wgt + (size_t)mc * Ktot + k0 + aK;
            if (k0 + aK + 8 <= Ktot) {
                if (k0 + 32 < Ktot) __builtin_prefetch(wr + 32, 0, 1);
                const float2* wr2 = (const float2*)wr;   // rows 8B-aligned (K even)
#pragma unroll
                for (int r = 0; r < 4; ++r) {
                    float2 u = wr2[r];
                    tmpA[2 * r] = u.x; tmpA[2 * r + 1] = u.y;
                }
            } else {
#pragma unroll
                for (int r = 0; r < 8; ++r) {
                    const int kg = k0 + aK + r;
                    const int kc = (kg < Ktot) ? kg : (Ktot - 1);
                    float vv = wgt[(size_t)mc * Ktot + kc];
                    tmpA[r] = (kg < Ktot) ? vv : 0.0f;
                }
            }
            v8h pa;
#pragma unroll
            for (int r = 0; r < 8; ++r) pa[r] = (_Float16)(tmpA[r] * fm);
            *(v8h*)(ldsA + aMl * 32 + aK) = pa;
        }

        // ================= stage B (im2col gather) =================
        {
            float tmpB[8];
#pragma unroll
            for (int r = 0; r < 8; ++r) {
                const int kg = k0 + bK + r;      // wave-uniform
                float v = 0.0f;
                if (kg < Ktot) {
                    const int ci = kg / khw;
                    const int rr = kg - ci * khw;
                    const int kh = rr / KW;
                    const int kw = rr - kh * KW;
                    const int iy = pyy + kh - ph;
                    const int ix = pxx + kw - pw;
                    const bool ok = (iy >= 0) & (iy < H_) & (ix >= 0) & (ix < W_);
                    const int iyc = ok ? iy : 0;
                    const int ixc = ok ? ix : 0;
                    const float* s; int cS; int cl = ci;
                    if (cl < c0)              { s = s0; cS = c0; }
                    else if ((cl -= c0) < c1) { s = s1; cS = c1; }
                    else if ((cl -= c1) < c2) { s = s2; cS = c2; }
                    else                      { cl -= c2; s = s3; cS = c3; }
                    v = s[((size_t)(nImg * cS + cl) * H_ + iyc) * W_ + ixc]
                        * (ok ? 1.0f : 0.0f);
                }
                tmpB[r] = v;
            }
            v8h pb;
#pragma unroll
            for (int r = 0; r < 8; ++r) pb[r] = (_Float16)tmpB[r];
            *(v8h*)(ldsB + bPx * 32 + bK) = pb;
        }

        __syncthreads();

        // ================= fragments + 2 WMMAs =================
        const int mrow = mSub * 16 + mn;
        v8h alo = *(const v8h*)(ldsA + mrow * 32 + laneHi * 16);
        v8h ahi = *(const v8h*)(ldsA + mrow * 32 + laneHi * 16 + 8);
        v16h a = __builtin_shufflevector(alo, ahi,
                     0,1,2,3,4,5,6,7,8,9,10,11,12,13,14,15);
        v16h b0 = *(const v16h*)(ldsB + (nSub * 32 + mn) * 32 + laneHi * 16);
        v16h b1 = *(const v16h*)(ldsB + (nSub * 32 + 16 + mn) * 32 + laneHi * 16);
        acc[0] = __builtin_amdgcn_wmma_f32_16x16x32_f16(
                     false, a, false, b0, (short)0, acc[0], false, false);
        acc[1] = __builtin_amdgcn_wmma_f32_16x16x32_f16(
                     false, a, false, b1, (short)0, acc[1], false, false);
    }

    // ---- epilogue: bias + fused activation
#pragma unroll
    for (int hb = 0; hb < 2; ++hb) {
        const int gout = nBase + nSub * 32 + hb * 16 + mn;
        const int nIo  = gout / HW_;
        const int po   = gout - nIo * HW_;
#pragma unroll
        for (int r = 0; r < 8; ++r) {
            const int mo = mBase + mSub * 16 + r + laneHi * 8;
            if (mo < Cout) {
                float v = acc[hb][r] + bias[mo];
                if      (act == ACT_RELU) v = fmaxf(v, 0.0f);
                else if (act == ACT_SIG)  v = 1.0f / (1.0f + __expf(-v));
                else if (act == ACT_TANH) v = tanhf(v);
                else if (act == ACT_QRT)  v *= 0.25f;
                out[(size_t)(nIo * Cout + mo) * HW_ + po] = v;
            }
        }
    }
}

// =====================================================================
// All-pairs correlation: pyr0[n,i,j] = sum_d F1[n,d,i]*F2[n,d,j] / 16
// Same 64x64 LDS-staged WMMA pipeline, K=256 (8 chunks), coalesced fills
// packed into one ds_store_b128 per thread per tile.
// =====================================================================
__global__ __launch_bounds__(256) void corr_wmma_f16(
    const float* __restrict__ f1, const float* __restrict__ f2,
    float* __restrict__ pyr0)
{
    __shared__ __attribute__((aligned(32))) _Float16 ldsA[64 * 32];  // [i][k]
    __shared__ __attribute__((aligned(32))) _Float16 ldsB[64 * 32];  // [j][k]

    const int t      = threadIdx.x;
    const int lane   = t & 31;
    const int wv     = t >> 5;
    const int mSub   = wv >> 1;
    const int nSub   = wv & 1;
    const int laneHi = lane >> 4;
    const int mn     = lane & 15;

    const int iBase = blockIdx.y * 64;
    const int jBase = blockIdx.x * 64;
    const int n     = blockIdx.z;

    const float* F1 = f1 + (size_t)n * DFEAT_ * HW_;
    const float* F2 = f2 + (size_t)n * DFEAT_ * HW_;

    const int pl = t & 63;          // pixel column owned for fills (coalesced)
    const int kO = (t >> 6) * 8;    // 8 consecutive k slots

    v8f acc[2] = {};
#pragma unroll 1
    for (int k0 = 0; k0 < DFEAT_; k0 += 32) {
        __syncthreads();
        {
            float ta[8], tb[8];
#pragma unroll
            for (int r = 0; r < 8; ++r)
                ta[r] = F1[(size_t)(k0 + kO + r) * HW_ + iBase + pl];
#pragma unroll
            for (int r = 0; r < 8; ++r)
                tb[r] = F2[(size_t)(k0 + kO + r) * HW_ + jBase + pl];
            v8h pa, pb;
#pragma unroll
            for (int r = 0; r < 8; ++r) { pa[r] = (_Float16)ta[r]; pb[r] = (_Float16)tb[r]; }
            *(v8h*)(ldsA + pl * 32 + kO) = pa;
            *(v8h*)(ldsB + pl * 32 + kO) = pb;
        }
        __syncthreads();

        const int mrow = mSub * 16 + mn;
        v8h alo = *(const v8h*)(ldsA + mrow * 32 + laneHi * 16);
        v8h ahi = *(const v8h*)(ldsA + mrow * 32 + laneHi * 16 + 8);
        v16h a = __builtin_shufflevector(alo, ahi,
                     0,1,2,3,4,5,6,7,8,9,10,11,12,13,14,15);
        v16h b0 = *(const v16h*)(ldsB + (nSub * 32 + mn) * 32 + laneHi * 16);
        v16h b1 = *(const v16h*)(ldsB + (nSub * 32 + 16 + mn) * 32 + laneHi * 16);
        acc[0] = __builtin_amdgcn_wmma_f32_16x16x32_f16(
                     false, a, false, b0, (short)0, acc[0], false, false);
        acc[1] = __builtin_amdgcn_wmma_f32_16x16x32_f16(
                     false, a, false, b1, (short)0, acc[1], false, false);
    }

#pragma unroll
    for (int hb = 0; hb < 2; ++hb) {
        const int jj = jBase + nSub * 32 + hb * 16 + mn;
#pragma unroll
        for (int r = 0; r < 8; ++r) {
            const int ii = iBase + mSub * 16 + r + laneHi * 8;
            pyr0[((size_t)(n * HW_ + ii)) * HW_ + jj] = acc[hb][r] * 0.0625f;
        }
    }
}

// ---------------- pyramid avg-pool 2x2 ----------------
__global__ void avgpool2x2(const float* __restrict__ in, float* __restrict__ out,
                           int Hin, int Win)
{
    const int Hout = Hin >> 1, Wout = Win >> 1;
    const long long total = (long long)NPIX_ * Hout * Wout;
    long long idx = (long long)blockIdx.x * blockDim.x + threadIdx.x;
    if (idx >= total) return;
    int x = (int)(idx % Wout); long long tt = idx / Wout;
    int y = (int)(tt % Hout);  long long b = tt / Hout;
    const float* ip = in + b * (long long)(Hin * Win);
    out[idx] = 0.25f * (ip[(2 * y) * Win + 2 * x]     + ip[(2 * y) * Win + 2 * x + 1] +
                        ip[(2 * y + 1) * Win + 2 * x] + ip[(2 * y + 1) * Win + 2 * x + 1]);
}

// ---------------- bilinear correlation lookup ----------------
__device__ __forceinline__ float bilin(const float* __restrict__ img,
                                       int Hl, int Wl, float sx, float sy)
{
    float x0f = floorf(sx), y0f = floorf(sy);
    int   x0  = (int)x0f,   y0  = (int)y0f;
    float wx1 = sx - x0f, wx0 = 1.0f - wx1;
    float wy1 = sy - y0f, wy0 = 1.0f - wy1;
    float acc = 0.0f;
#define GETV(yi, xi) (((xi) >= 0 && (xi) < Wl && (yi) >= 0 && (yi) < Hl) ? img[(yi) * Wl + (xi)] : 0.0f)
    acc += GETV(y0,     x0)     * (wy0 * wx0);
    acc += GETV(y0,     x0 + 1) * (wy0 * wx1);
    acc += GETV(y0 + 1, x0)     * (wy1 * wx0);
    acc += GETV(y0 + 1, x0 + 1) * (wy1 * wx1);
#undef GETV
    return acc;
}

__global__ void corr_lookup_k(const float* __restrict__ coords1,
                              const float* __restrict__ p0, const float* __restrict__ p1,
                              const float* __restrict__ p2, const float* __restrict__ p3,
                              float* __restrict__ corr)
{
    const long long total = (long long)NPIX_ * 324;
    long long idx = (long long)blockIdx.x * blockDim.x + threadIdx.x;
    if (idx >= total) return;
    int ch = (int)(idx % 324); long long b = idx / 324;
    int n = (int)(b / HW_);
    int p = (int)(b - (long long)n * HW_);
    int lvl = ch / 81;
    int cc  = ch - lvl * 81;
    int ai  = cc / 9, bi = cc - ai * 9;   // delta = meshgrid(dy,dx) quirk: x+=d[ai], y+=d[bi]
    float cx = coords1[((size_t)(n * 2 + 0)) * HW_ + p];
    float cy = coords1[((size_t)(n * 2 + 1)) * HW_ + p];
    float scale = 1.0f / (float)(1 << lvl);
    float sx = cx * scale + (float)(ai - 4);
    float sy = cy * scale + (float)(bi - 4);
    const int Hl = H_ >> lvl, Wl = W_ >> lvl;
    const float* base = (lvl == 0) ? p0 : (lvl == 1) ? p1 : (lvl == 2) ? p2 : p3;
    const float* img  = base + b * (long long)(Hl * Wl);
    corr[((size_t)(n * 324 + ch)) * HW_ + p] = bilin(img, Hl, Wl, sx, sy);
}

// ---------------- pointwise helpers ----------------
__global__ void k_tanh(const float* __restrict__ in, float* __restrict__ out, int n)
{ int i = blockIdx.x * blockDim.x + threadIdx.x; if (i < n) out[i] = tanhf(in[i]); }

__global__ void k_relu(const float* __restrict__ in, float* __restrict__ out, int n)
{ int i = blockIdx.x * blockDim.x + threadIdx.x; if (i < n) out[i] = fmaxf(in[i], 0.0f); }

__global__ void k_coords_init(float* __restrict__ c)
{
    int idx = blockIdx.x * blockDim.x + threadIdx.x;
    if (idx >= NPIX_) return;
    int n = idx / HW_; int p = idx - n * HW_;
    c[((size_t)(n * 2 + 0)) * HW_ + p] = (float)(p % W_);
    c[((size_t)(n * 2 + 1)) * HW_ + p] = (float)(p / W_);
}

__global__ void k_flow(const float* __restrict__ c, float* __restrict__ fl)
{
    int idx = blockIdx.x * blockDim.x + threadIdx.x;
    if (idx >= NPIX_) return;
    int n = idx / HW_; int p = idx - n * HW_;
    fl[((size_t)(n * 2 + 0)) * HW_ + p] = c[((size_t)(n * 2 + 0)) * HW_ + p] - (float)(p % W_);
    fl[((size_t)(n * 2 + 1)) * HW_ + p] = c[((size_t)(n * 2 + 1)) * HW_ + p] - (float)(p / W_);
}

__global__ void k_mul(const float* __restrict__ a, const float* __restrict__ b,
                      float* __restrict__ o, int n)
{ int i = blockIdx.x * blockDim.x + threadIdx.x; if (i < n) o[i] = a[i] * b[i]; }

__global__ void k_gru(const float* __restrict__ z, const float* __restrict__ q,
                      float* __restrict__ net, int n)
{
    int i = blockIdx.x * blockDim.x + threadIdx.x;
    if (i < n) { float zz = z[i]; net[i] = (1.0f - zz) * net[i] + zz * q[i]; }
}

__global__ void k_add_inplace(float* __restrict__ a, const float* __restrict__ b, int n)
{ int i = blockIdx.x * blockDim.x + threadIdx.x; if (i < n) a[i] += b[i]; }

// ---------------- convex upsampler (softmax over 9 taps) ----------------
__global__ void upsample_k(const float* __restrict__ coords1,
                           const float* __restrict__ mask,
                           float* __restrict__ out)
{
    const int total = NB_ * 2 * (8 * H_) * (8 * W_);
    int idx = blockIdx.x * blockDim.x + threadIdx.x;
    if (idx >= total) return;
    int ox = idx % (8 * W_); int t = idx / (8 * W_);
    int oy = t % (8 * H_);   t /= (8 * H_);
    int c = t % 2; int n = t / 2;
    int x = ox >> 3, j8 = ox & 7;
    int y = oy >> 3, i8 = oy & 7;

    float mv[9]; float mx = -1e30f;
#pragma unroll
    for (int k = 0; k < 9; ++k) {
        mv[k] = mask[((size_t)(n * 576 + k * 64 + i8 * 8 + j8)) * HW_ + y * W_ + x];
        mx = fmaxf(mx, mv[k]);
    }
    float ssum = 0.0f;
#pragma unroll
    for (int k = 0; k < 9; ++k) { mv[k] = __expf(mv[k] - mx); ssum += mv[k]; }
    float r = 0.0f;
#pragma unroll
    for (int k = 0; k < 9; ++k) {
        int dy = k / 3 - 1, dx = (k - (k / 3) * 3) - 1;
        int yy = y + dy, xx = x + dx;
        float fv = 0.0f;
        if (yy >= 0 && yy < H_ && xx >= 0 && xx < W_) {
            float ref = (c == 0) ? (float)xx : (float)yy;   // flow = coords1 - coords0
            fv = 8.0f * (coords1[((size_t)(n * 2 + c)) * HW_ + yy * W_ + xx] - ref);
        }
        r += mv[k] * fv;
    }
    out[idx] = r / ssum;
}

// =====================================================================
// host driver
// =====================================================================
extern "C" void kernel_launch(void* const* d_in, const int* in_sizes, int n_in,
                              void* d_out, int out_size, void* d_ws, size_t ws_size,
                              hipStream_t stream)
{
    (void)in_sizes; (void)n_in; (void)out_size; (void)ws_size;

    const float* fmap1  = (const float*)d_in[0];
    const float* fmap2  = (const float*)d_in[1];
    const float* net_in = (const float*)d_in[2];
    const float* inp_in = (const float*)d_in[3];
    const float* W[15]; const float* Bi[15];
    for (int l = 0; l < 15; ++l) { W[l] = (const float*)d_in[4 + 2 * l]; Bi[l] = (const float*)d_in[5 + 2 * l]; }
    enum { L_MC1, L_MC2, L_MF1, L_MF2, L_MCONV,
           L_GZ1, L_GR1, L_GQ1, L_GZ2, L_GR2, L_GQ2,
           L_FH1, L_FH2, L_MK1, L_MK2 };

    float* ws = (float*)d_ws;
    size_t cur = 0;
    auto alloc = [&](size_t nf) { float* p = ws + cur; cur += nf; return p; };

    float* coords1 = alloc((size_t)NB_ * 2   * HW_);
    float* net     = alloc((size_t)NB_ * 128 * HW_);
    float* inp     = alloc((size_t)NB_ * 128 * HW_);
    float* corrb   = alloc((size_t)NB_ * 324 * HW_);
    float* cor1    = alloc((size_t)NB_ * 256 * HW_);
    float* cor2    = alloc((size_t)NB_ * 192 * HW_);
    float* flo1    = alloc((size_t)NB_ * 128 * HW_);
    float* flo2    = alloc((size_t)NB_ * 64  * HW_);
    float* moutb   = alloc((size_t)NB_ * 126 * HW_);
    float* flowb   = alloc((size_t)NB_ * 2   * HW_);
    float* zb      = alloc((size_t)NB_ * 128 * HW_);
    float* rb      = alloc((size_t)NB_ * 128 * HW_);
    float* qb      = alloc((size_t)NB_ * 128 * HW_);
    float* rnet    = alloc((size_t)NB_ * 128 * HW_);
    float* t1      = alloc((size_t)NB_ * 256 * HW_);
    float* dflow   = alloc((size_t)NB_ * 2   * HW_);
    float* t2      = alloc((size_t)NB_ * 256 * HW_);
    float* upm     = alloc((size_t)NB_ * 576 * HW_);
    float* pyr0    = alloc((size_t)NPIX_ * H_ * W_);
    float* pyr1    = alloc((size_t)NPIX_ * (H_ / 2) * (W_ / 2));
    float* pyr2    = alloc((size_t)NPIX_ * (H_ / 4) * (W_ / 4));
    float* pyr3    = alloc((size_t)NPIX_ * (H_ / 8) * (W_ / 8));

    const int PT = 256;
    auto nb = [&](long long n) { return (unsigned)((n + PT - 1) / PT); };

    auto conv = [&](int layer,
                    const float* a0, int ca, const float* a1, int cb,
                    const float* a2, int cc, const float* a3, int cd,
                    float* op, int Cout, int KH, int KW, int act) {
        dim3 grid(NPIX_ / 64, (Cout + 63) / 64, 1);
        conv2d_wmma_f16<<<grid, 256, 0, stream>>>(W[layer], Bi[layer],
            a0, ca, a1, cb, a2, cc, a3, cd, op, Cout, KH, KW, act);
    };

    const int NC128 = NB_ * 128 * HW_;

    // ---- init state + correlation pyramid ----
    k_tanh<<<nb(NC128), PT, 0, stream>>>(net_in, net, NC128);
    k_relu<<<nb(NC128), PT, 0, stream>>>(inp_in, inp, NC128);
    k_coords_init<<<nb(NPIX_), PT, 0, stream>>>(coords1);

    corr_wmma_f16<<<dim3(HW_ / 64, HW_ / 64, NB_), 256, 0, stream>>>(fmap1, fmap2, pyr0);
    avgpool2x2<<<nb((long long)NPIX_ * 24 * 32), PT, 0, stream>>>(pyr0, pyr1, 48, 64);
    avgpool2x2<<<nb((long long)NPIX_ * 12 * 16), PT, 0, stream>>>(pyr1, pyr2, 24, 32);
    avgpool2x2<<<nb((long long)NPIX_ * 6 * 8),   PT, 0, stream>>>(pyr2, pyr3, 12, 16);

    // ---- iterative updates ----
    for (int it = 0; it < ITERS_; ++it) {
        corr_lookup_k<<<nb((long long)NPIX_ * 324), PT, 0, stream>>>(
            coords1, pyr0, pyr1, pyr2, pyr3, corrb);
        k_flow<<<nb(NPIX_), PT, 0, stream>>>(coords1, flowb);

        // motion encoder
        conv(L_MC1,  corrb, 324, nullptr, 0, nullptr, 0, nullptr, 0, cor1, 256, 1, 1, ACT_RELU);
        conv(L_MC2,  cor1,  256, nullptr, 0, nullptr, 0, nullptr, 0, cor2, 192, 3, 3, ACT_RELU);
        conv(L_MF1,  flowb, 2,   nullptr, 0, nullptr, 0, nullptr, 0, flo1, 128, 7, 7, ACT_RELU);
        conv(L_MF2,  flo1,  128, nullptr, 0, nullptr, 0, nullptr, 0, flo2, 64,  3, 3, ACT_RELU);
        conv(L_MCONV, cor2, 192, flo2, 64, nullptr, 0, nullptr, 0,  moutb, 126, 3, 3, ACT_RELU);

        // horizontal (1x5) GRU — hx = [net | inp | mout | flow] via concat-view
        conv(L_GZ1, net,  128, inp, 128, moutb, 126, flowb, 2, zb, 128, 1, 5, ACT_SIG);
        conv(L_GR1, net,  128, inp, 128, moutb, 126, flowb, 2, rb, 128, 1, 5, ACT_SIG);
        k_mul<<<nb(NC128), PT, 0, stream>>>(rb, net, rnet, NC128);
        conv(L_GQ1, rnet, 128, inp, 128, moutb, 126, flowb, 2, qb, 128, 1, 5, ACT_TANH);
        k_gru<<<nb(NC128), PT, 0, stream>>>(zb, qb, net, NC128);

        // vertical (5x1) GRU
        conv(L_GZ2, net,  128, inp, 128, moutb, 126, flowb, 2, zb, 128, 5, 1, ACT_SIG);
        conv(L_GR2, net,  128, inp, 128, moutb, 126, flowb, 2, rb, 128, 5, 1, ACT_SIG);
        k_mul<<<nb(NC128), PT, 0, stream>>>(rb, net, rnet, NC128);
        conv(L_GQ2, rnet, 128, inp, 128, moutb, 126, flowb, 2, qb, 128, 5, 1, ACT_TANH);
        k_gru<<<nb(NC128), PT, 0, stream>>>(zb, qb, net, NC128);

        // flow head + mask head
        conv(L_FH1, net, 128, nullptr, 0, nullptr, 0, nullptr, 0, t1,    256, 3, 3, ACT_RELU);
        conv(L_FH2, t1,  256, nullptr, 0, nullptr, 0, nullptr, 0, dflow, 2,   3, 3, ACT_NONE);
        conv(L_MK1, net, 128, nullptr, 0, nullptr, 0, nullptr, 0, t2,    256, 3, 3, ACT_RELU);
        conv(L_MK2, t2,  256, nullptr, 0, nullptr, 0, nullptr, 0, upm,   576, 1, 1, ACT_QRT);

        k_add_inplace<<<nb(NB_ * 2 * HW_), PT, 0, stream>>>(coords1, dflow, NB_ * 2 * HW_);

        const size_t predStride = (size_t)NB_ * 2 * (8 * H_) * (8 * W_);  // 786432
        upsample_k<<<nb((long long)predStride), PT, 0, stream>>>(
            coords1, upm, (float*)d_out + (size_t)it * predStride);
    }
}
